// HamiltonianNeuralODE_16801912062408
// MI455X (gfx1250) — matrix operand order repaired and checked by reading, not compiled, not run on hardware
//
#include <hip/hip_runtime.h>

// ---------------------------------------------------------------------------
// Hamiltonian Neural ODE (leapfrog, separable H) on MI455X / gfx1250.
//   grad_V(q) = W1 * ((1 - tanh^2(W1^T q + b1)) .* w2)   (per token, D=256, HID=1024)
// Tokens are independent -> each workgroup integrates its 32-token tile for all
// 10 steps locally (no inter-block sync). Matmuls use v_wmma_f32_16x16x32_bf16
// with f32 accumulation; weights are pre-converted to bf16 in both [d][h] and
// [h][d] layouts so every WMMA B fragment is a contiguous 2 x b128 LDS read.
// ---------------------------------------------------------------------------

typedef __attribute__((ext_vector_type(16))) __bf16 v16bf;
typedef __attribute__((ext_vector_type(8)))  float  v8f;
typedef __attribute__((ext_vector_type(4)))  unsigned int u32x4;

union Frag { v16bf v; u32x4 q[2]; };

#define D_MODEL 256
#define HID     1024
#define MT      32      // tokens per workgroup
#define CH      64      // hidden-dim chunk
#define NEVALS  11      // 1 + NUM_STEPS gradient evaluations
#define DT_F    0.1f

static __device__ __forceinline__ unsigned short f2bf(float f) {
    unsigned int u = __float_as_uint(f);
    u += 0x7FFFu + ((u >> 16) & 1u);        // round-to-nearest-even
    return (unsigned short)(u >> 16);
}

static __device__ __forceinline__ v8f wmma_bf16(v16bf a, v16bf b, v8f c) {
    // (neg_a, A, neg_b, B, c_mod, C, reuse_a, reuse_b)
    return __builtin_amdgcn_wmma_f32_16x16x32_bf16(false, a, false, b,
                                                   (short)0, c, false, false);
}

// Prep: W1 (f32, 256x1024 row-major) -> bf16 in both layouts.
//   Wb[d*HID + h] = bf16(W1[d][h])   (B frags for grad = G * W1^T)
//   Wt[h*D   + d] = bf16(W1[d][h])   (B frags for Z    = Q * W1)
__global__ __launch_bounds__(256) void hamnode_prep(const float* __restrict__ W1,
                                                    unsigned short* __restrict__ Wt,
                                                    unsigned short* __restrict__ Wb) {
    int idx = blockIdx.x * 256 + threadIdx.x;       // 0 .. 256*1024-1
    if (idx < D_MODEL * HID) {
        int d = idx >> 10;
        int h = idx & (HID - 1);
        unsigned short v = f2bf(W1[idx]);
        Wb[idx] = v;
        Wt[h * D_MODEL + d] = v;
    }
}

__global__ __launch_bounds__(256) void hamnode_leapfrog(
        const float* __restrict__ x,       // (16384, 512) = [q | p]
        const float* __restrict__ b1,      // (1024)
        const float* __restrict__ w2,      // (1024)
        const unsigned short* __restrict__ Wt,   // bf16 [h][d]
        const unsigned short* __restrict__ Wb,   // bf16 [d][h]
        float* __restrict__ out) {

    __shared__ float          qf[MT * D_MODEL];   // 32 KB  f32 master q
    __shared__ float          pf[MT * D_MODEL];   // 32 KB  f32 master p
    __shared__ unsigned short qb[MT * D_MODEL];   // 16 KB  bf16 q (A frags, GEMM1)
    __shared__ unsigned short wt1[CH * D_MODEL];  // 32 KB  Wt chunk [h_loc][d]
    __shared__ unsigned short wt2[D_MODEL * CH];  // 32 KB  Wb chunk [d][h_loc]
    __shared__ unsigned short gbuf[MT * CH];      //  4 KB  G chunk  [m][h_loc]
    __shared__ float          sb1[HID];           //  4 KB
    __shared__ float          sw2[HID];           //  4 KB   total ~156 KB LDS

    const int tid  = threadIdx.x;
    const int wave = tid >> 5;
    const int lane = tid & 31;
    const int half = lane >> 4;       // lane group (ISA §7.12.2 layouts)
    const int l16  = lane & 15;
    const int rw   = wave & 1;        // token-row group (16 rows)
    const int cw   = wave >> 1;       // column group (0..3)
    const int rb   = rw * 16;
    const long t0  = (long)blockIdx.x * MT;

    // ---- load tile state + bias/w2 cache -----------------------------------
    for (int i = tid; i < HID; i += 256) { sb1[i] = b1[i]; sw2[i] = w2[i]; }
    for (int i = tid; i < MT * D_MODEL; i += 256) {
        int m = i >> 8, d = i & 255;
        long base = (t0 + m) * (2 * D_MODEL);
        qf[i] = x[base + d];
        pf[i] = x[base + D_MODEL + d];
    }
    __syncthreads();

    v8f acc2[4];

    for (int e = 0; e < NEVALS; ++e) {
        // bf16 snapshot of q for this gradient evaluation
        for (int i = tid; i < MT * D_MODEL; i += 256) qb[i] = f2bf(qf[i]);
        __syncthreads();

        #pragma unroll
        for (int t4 = 0; t4 < 4; ++t4) acc2[t4] = (v8f)0.0f;

        for (int c = 0; c < 16; ++c) {
            const int h0 = c * CH;
            __syncthreads();   // previous chunk's wt/gbuf readers are done

            // stage weight chunks into LDS (contiguous 32 KB + 256 strided rows)
            {
                const u32x4* s1 = reinterpret_cast<const u32x4*>(Wt + h0 * D_MODEL);
                u32x4*       d1 = reinterpret_cast<u32x4*>(wt1);
                #pragma unroll 4
                for (int i = tid; i < (CH * D_MODEL) / 8; i += 256) d1[i] = s1[i];

                const u32x4* s2 = reinterpret_cast<const u32x4*>(Wb + (long)tid * HID + h0);
                u32x4*       d2 = reinterpret_cast<u32x4*>(wt2 + tid * CH);
                #pragma unroll
                for (int j = 0; j < 8; ++j) d2[j] = s2[j];

                if (c + 1 < 16)   // hint next chunk into cache (global_prefetch_b8)
                    __builtin_prefetch((const void*)(Wt + (h0 + CH) * D_MODEL + tid * 16), 0, 1);
            }
            __syncthreads();

            // ---- GEMM1: Z(32xCH) = Q(32x256) @ W1[:,chunk]; one 16x16 tile/wave
            v8f z = (v8f)0.0f;
            {
                const int am = rb + l16;           // A: token row
                const int bn = cw * 16 + l16;      // B: local hidden col (row of wt1)
                #pragma unroll
                for (int kk = 0; kk < 8; ++kk) {
                    const int k0 = kk * 32;
                    Frag a, b;
                    a.q[0] = *reinterpret_cast<const u32x4*>(qb + am * D_MODEL + k0 + (half ? 16 : 0));
                    a.q[1] = *reinterpret_cast<const u32x4*>(qb + am * D_MODEL + k0 + (half ? 16 : 0) + 8);
                    const int ko = k0 + (half ? 16 : 0);
                    b.q[0] = *reinterpret_cast<const u32x4*>(wt1 + bn * D_MODEL + ko);
                    b.q[1] = *reinterpret_cast<const u32x4*>(wt1 + bn * D_MODEL + ko + 8);
                    z = wmma_bf16(a.v, b.v, z);
                }
            }

            // ---- activation: g = (1 - tanh^2(z + b1)) * w2, write bf16 G chunk
            {
                const int hloc = cw * 16 + l16;
                const int h    = h0 + hloc;
                const float bb = sb1[h], ww = sw2[h];
                #pragma unroll
                for (int i = 0; i < 8; ++i) {
                    const int m = rb + i + 8 * half;      // C-layout row (ISA §7.12.2)
                    float zz = z[i] + bb;
                    float ex = __expf(2.0f * zz);
                    float t  = 1.0f - 2.0f / (ex + 1.0f); // tanh
                    float g  = (1.0f - t * t) * ww;
                    gbuf[m * CH + hloc] = f2bf(g);
                }
            }
            __syncthreads();

            // ---- GEMM2: grad(32x256) += G(32xCH) @ W1t[chunk,:]; 4 tiles/wave
            #pragma unroll
            for (int kk = 0; kk < 2; ++kk) {
                const int k0 = kk * 32;
                Frag a;
                a.q[0] = *reinterpret_cast<const u32x4*>(gbuf + (rb + l16) * CH + k0 + (half ? 16 : 0));
                a.q[1] = *reinterpret_cast<const u32x4*>(gbuf + (rb + l16) * CH + k0 + (half ? 16 : 0) + 8);
                #pragma unroll
                for (int t4 = 0; t4 < 4; ++t4) {
                    const int dcol = cw * 64 + t4 * 16 + l16;   // output d col = wt2 row
                    const int ko   = k0 + (half ? 16 : 0);
                    Frag b;
                    b.q[0] = *reinterpret_cast<const u32x4*>(wt2 + dcol * CH + ko);
                    b.q[1] = *reinterpret_cast<const u32x4*>(wt2 + dcol * CH + ko + 8);
                    acc2[t4] = wmma_bf16(a.v, b.v, acc2[t4]);
                }
            }
        } // chunks

        // ---- half kick(s): e==0 and e==10 get one 0.5*dt kick, middle evals
        // get two (matching the reference's two separate subtractions).
        {
            const float sc = 0.5f * DT_F;
            const int nk = (e == 0 || e == NEVALS - 1) ? 1 : 2;
            #pragma unroll
            for (int t4 = 0; t4 < 4; ++t4) {
                #pragma unroll
                for (int i = 0; i < 8; ++i) {
                    const int m = rb + i + 8 * half;
                    const int d = cw * 64 + t4 * 16 + l16;
                    float pv = pf[m * D_MODEL + d];
                    pv -= sc * acc2[t4][i];
                    if (nk == 2) pv -= sc * acc2[t4][i];
                    pf[m * D_MODEL + d] = pv;
                }
            }
        }
        __syncthreads();

        if (e < NEVALS - 1) {                 // drift: q += dt * p_half
            for (int i = tid; i < MT * D_MODEL; i += 256) qf[i] += DT_F * pf[i];
            __syncthreads();
        }
    }

    // ---- write back ---------------------------------------------------------
    for (int i = tid; i < MT * D_MODEL; i += 256) {
        int m = i >> 8, d = i & 255;
        long base = (t0 + m) * (2 * D_MODEL);
        out[base + d]           = qf[i];
        out[base + D_MODEL + d] = pf[i];
    }
}

extern "C" void kernel_launch(void* const* d_in, const int* in_sizes, int n_in,
                              void* d_out, int out_size, void* d_ws, size_t ws_size,
                              hipStream_t stream) {
    const float* x  = (const float*)d_in[0];   // (8,2048,512)
    const float* W1 = (const float*)d_in[1];   // (256,1024)
    const float* b1 = (const float*)d_in[2];   // (1024,)
    const float* W2 = (const float*)d_in[3];   // (1024,1)
    // d_in[4] = b2: constant offset on V, does not affect grad_V -> unused.

    unsigned short* Wt = (unsigned short*)d_ws;          // 512 KB  [h][d]
    unsigned short* Wb = Wt + (size_t)HID * D_MODEL;     // 512 KB  [d][h]

    hamnode_prep<<<(D_MODEL * HID + 255) / 256, 256, 0, stream>>>(W1, Wt, Wb);

    const int n_tokens = 8 * 2048;
    hamnode_leapfrog<<<n_tokens / MT, 256, 0, stream>>>(x, b1, W2, Wt, Wb,
                                                        (float*)d_out);
}
